// DeformableNet_24876450579397
// MI455X (gfx1250) — compile-verified, matrix operand order checked by reading, not run on hardware
//
#include <hip/hip_runtime.h>
#include <hip/hip_bf16.h>
#include <stdint.h>

typedef __attribute__((ext_vector_type(16))) _Float16 v16h;
typedef __attribute__((ext_vector_type(8)))  _Float16 v8h;
typedef __attribute__((ext_vector_type(8)))  float    v8f;
typedef __attribute__((ext_vector_type(4)))  int      v4i;

#define BN_SCALE 0.99950037f
#define ALPHA    0.2f

#if defined(__has_builtin)
#  if __has_builtin(__builtin_amdgcn_global_load_async_to_lds_b128)
#    define HAVE_ASYNC_LDS 1
#  endif
#endif
#ifndef HAVE_ASYNC_LDS
#  define HAVE_ASYNC_LDS 0
#endif

// address-space casts for the async global->LDS builtin
#define AS1_V4I(p) ((__attribute__((address_space(1))) v4i*)(uintptr_t)(p))
#define AS3_V4I(p) ((__attribute__((address_space(3))) v4i*)(uint32_t)(uintptr_t)(p))

// lrelu(x) = x>0 ? x : a*x  ==  max(x, a*x)   (a in (0,1))
__device__ __forceinline__ float lrelu(float x)    { return fmaxf(x, ALPHA * x); }
__device__ __forceinline__ float lrelu_bn(float x) { float y = x * BN_SCALE; return fmaxf(y, ALPHA * y); }

// ---------------------------------------------------------------------------
// Weight packing: HWIO f32 [taps][32][32] -> WMMA B-fragment f16
// layout [tap][half][lane(32)][slot(16)].
// K bit map (mirrors 16-bit A/B VGPR layout): k[2:0]=slot[2:0], k[3]=lane[4],
// k[4]=slot[3].  N = (lane&15) + 16*half.
// ---------------------------------------------------------------------------
__global__ void pack_weights(const float* __restrict__ w, _Float16* __restrict__ out,
                             int taps) {
    int idx = blockIdx.x * blockDim.x + threadIdx.x;
    int total = taps * 2 * 32 * 16;
    if (idx >= total) return;
    int slot =  idx        & 15;
    int lane = (idx >> 4)  & 31;
    int half = (idx >> 9)  & 1;
    int t    =  idx >> 10;
    int k = (slot & 7) | (((lane >> 4) & 1) << 3) | ((slot >> 3) << 4);
    int n = (lane & 15) + 16 * half;
    out[idx] = (_Float16)w[(t * 32 + k) * 32 + n];
}

// ---------------------------------------------------------------------------
// Stage 1: concat(fixed,moving) -> conv3x3 (2->32) -> BN -> lrelu -> avgpool2
// in:  fixed/moving f32 [4,512,512,1];  out: f16 NHWC [4,256,256,32]
// ---------------------------------------------------------------------------
__global__ __launch_bounds__(256) void conv1_pool(
        const float* __restrict__ fixed, const float* __restrict__ moving,
        const float* __restrict__ w0, _Float16* __restrict__ out) {
    __shared__ float ws[3][3][2][32];                 // 576 floats
    for (int i = threadIdx.x; i < 576; i += 256) ((float*)ws)[i] = w0[i];
    __syncthreads();

    int idx = blockIdx.x * blockDim.x + threadIdx.x;  // over 4*256*256
    if (idx >= 4 * 256 * 256) return;
    int xo = idx & 255, yo = (idx >> 8) & 255, b = idx >> 16;

    float fx[4][4], mv[4][4];
    #pragma unroll
    for (int wy = 0; wy < 4; ++wy) {
        int iy = 2 * yo - 1 + wy;
        #pragma unroll
        for (int wx = 0; wx < 4; ++wx) {
            int ix = 2 * xo - 1 + wx;
            bool ok = (unsigned)iy < 512u && (unsigned)ix < 512u;
            size_t p = ((size_t)(b * 512 + (ok ? iy : 0)) * 512) + (ok ? ix : 0);
            fx[wy][wx] = ok ? fixed[p]  : 0.0f;
            mv[wy][wx] = ok ? moving[p] : 0.0f;
        }
    }
    __align__(16) _Float16 res[32];
    for (int oc = 0; oc < 32; ++oc) {
        float pooled = 0.0f;
        #pragma unroll
        for (int dy = 0; dy < 2; ++dy)
        #pragma unroll
        for (int dx = 0; dx < 2; ++dx) {
            float c = 0.0f;
            #pragma unroll
            for (int ky = 0; ky < 3; ++ky)
            #pragma unroll
            for (int kx = 0; kx < 3; ++kx)
                c += fx[dy + ky][dx + kx] * ws[ky][kx][0][oc]
                   + mv[dy + ky][dx + kx] * ws[ky][kx][1][oc];
            pooled += 0.25f * lrelu_bn(c);
        }
        res[oc] = (_Float16)pooled;
    }
    v4i* o4 = (v4i*)(out + (((size_t)((b * 256 + yo) * 256 + xo)) << 5));
    #pragma unroll
    for (int i = 0; i < 4; ++i) o4[i] = ((const v4i*)res)[i];
}

// ---------------------------------------------------------------------------
// WMMA conv3x3 32->32, BN + lrelu, optional avgpool2.
// in f16 NHWC [B,H,W,32]; wpack f16 [9][2][32][16]; out f16 NHWC.
// Workgroup: 16x16 pixel tile, 8 waves.  The 18x18x32 halo tile is staged in
// LDS (async global->LDS when available); A fragments come from ds_load_b128,
// B fragments are loaded once per tap and reused across both row fragments.
// ---------------------------------------------------------------------------
template<bool POOL>
__global__ __launch_bounds__(256) void conv3x3_wmma(
        const _Float16* __restrict__ in, const _Float16* __restrict__ wpack,
        _Float16* __restrict__ out, int H, int W) {
    __shared__ v16h wl[9][2][32];                       // 18 KB packed B frags
    __shared__ __align__(16) _Float16 tile[18][18][32]; // 20.25 KB halo tile

    // weights -> LDS as b128 copies (9*2*32 v16h = 1152 v4i)
    {
        const v4i* src = (const v4i*)wpack;
        v4i* dst = (v4i*)wl;
        for (int i = threadIdx.x; i < 1152; i += 256) dst[i] = src[i];
    }

    const int b     = blockIdx.z;
    const int xbase = blockIdx.x * 16;
    const int ybase = blockIdx.y * 16;

    // input halo tile -> LDS: 324 pixels x 4 chunks of 16B
    for (int i = threadIdx.x; i < 324 * 4; i += 256) {
        int chunk = i & 3, pix = i >> 2;
        int tx = pix % 18, ty = pix / 18;
        int gx = xbase - 1 + tx, gy = ybase - 1 + ty;
        bool ok = (unsigned)gx < (unsigned)W && (unsigned)gy < (unsigned)H;
        _Float16* dstp = &tile[ty][tx][chunk * 8];
        const v4i* srcp = (const v4i*)(in +
            ((((size_t)b * H + (ok ? gy : 0)) * W + (ok ? gx : 0)) << 5)) + chunk;
#if HAVE_ASYNC_LDS
        if (ok) {
            __builtin_amdgcn_global_load_async_to_lds_b128(
                AS1_V4I(srcp), AS3_V4I(dstp), 0, 0);
        } else {
            v4i z = {0, 0, 0, 0};
            *(v4i*)dstp = z;
        }
#else
        v4i z = {0, 0, 0, 0};
        *(v4i*)dstp = ok ? *srcp : z;
#endif
    }
#if HAVE_ASYNC_LDS
#  if __has_builtin(__builtin_amdgcn_s_wait_asynccnt)
    __builtin_amdgcn_s_wait_asynccnt(0);
#  else
    asm volatile("s_wait_asynccnt 0" ::: "memory");
#  endif
#endif
    __syncthreads();

    const int lane = threadIdx.x & 31;
    const int wave = threadIdx.x >> 5;
    const int mcol = (lane & 15) + 1;       // A-row m = lane&15 -> tile x
    const int c0   = (lane >> 4) * 8;       // K chunk base per lane

    v8f acc0[2], acc1[2];
    acc0[0] = {}; acc0[1] = {}; acc1[0] = {}; acc1[1] = {};
    #pragma unroll
    for (int t = 0; t < 9; ++t) {
        const int ky = t / 3 - 1, kx = t % 3 - 1;
        v16h b0 = wl[t][0][lane];           // B loaded once per tap,
        v16h b1 = wl[t][1][lane];           // reused across both rows
        #pragma unroll
        for (int r = 0; r < 2; ++r) {
            const _Float16* p = &tile[2 * wave + r + 1 + ky][mcol + kx][c0];
            v8h lo = *(const v8h*)p;
            v8h hi = *(const v8h*)(p + 16);
            v16h a = __builtin_shufflevector(lo, hi, 0,1,2,3,4,5,6,7,
                                                     8,9,10,11,12,13,14,15);
            acc0[r] = __builtin_amdgcn_wmma_f32_16x16x32_f16(
                          false, a, false, b0, (short)0, acc0[r], false, false);
            acc1[r] = __builtin_amdgcn_wmma_f32_16x16x32_f16(
                          false, a, false, b1, (short)0, acc1[r], false, false);
        }
    }
    #pragma unroll
    for (int r = 0; r < 2; ++r)
    #pragma unroll
    for (int i = 0; i < 8; ++i) {
        acc0[r][i] = lrelu_bn(acc0[r][i]);
        acc1[r][i] = lrelu_bn(acc1[r][i]);
    }

    const int n0 = lane & 15;               // D: N = lane&15
    if (POOL) {
        const int Ho = H >> 1, Wo = W >> 1;
        const int yo = (ybase >> 1) + wave;
        #pragma unroll
        for (int p = 0; p < 4; ++p) {
            float s0 = 0.25f * (acc0[0][2*p] + acc0[0][2*p+1] +
                                acc0[1][2*p] + acc0[1][2*p+1]);
            float s1 = 0.25f * (acc1[0][2*p] + acc1[0][2*p+1] +
                                acc1[1][2*p] + acc1[1][2*p+1]);
            int xo = (xbase >> 1) + 4 * (lane >> 4) + p;
            size_t o = ((size_t)((b * Ho + yo) * Wo + xo)) << 5;
            out[o + n0]      = (_Float16)s0;
            out[o + n0 + 16] = (_Float16)s1;
        }
    } else {
        #pragma unroll
        for (int r = 0; r < 2; ++r) {
            int y = ybase + 2 * wave + r;
            #pragma unroll
            for (int i = 0; i < 8; ++i) {
                int x = xbase + 8 * (lane >> 4) + i;  // D: M = i + 8*lane[4]
                size_t o = ((size_t)((b * H + y) * W + x)) << 5;
                out[o + n0]      = (_Float16)acc0[r][i];
                out[o + n0 + 16] = (_Float16)acc1[r][i];
            }
        }
    }
}

// ---------------------------------------------------------------------------
// Pointwise head: pw0 (32->32, lrelu, WMMA) then pw1 (32->2) -> disp f32
// in f16 [4*64*64, 32]; wpack_pw0 [1][2][32][16]; pw1 f32 [32][2]
// ---------------------------------------------------------------------------
__global__ __launch_bounds__(256) void pointwise_wmma(
        const _Float16* __restrict__ in, const _Float16* __restrict__ wpack,
        const float* __restrict__ pw1, float* __restrict__ disp) {
    __shared__ v16h wb[2][32];                        // 2 KB
    __shared__ float stage[8][16][32];                // 16 KB
    {
        const v4i* src = (const v4i*)wpack;
        v4i* dst = (v4i*)wb;
        for (int i = threadIdx.x; i < 128; i += 256) dst[i] = src[i];
    }
    __syncthreads();

    const int lane = threadIdx.x & 31;
    const int wave = threadIdx.x >> 5;
    const int pixbase = blockIdx.x * 128 + wave * 16;

    const _Float16* p = in + (((size_t)(pixbase + (lane & 15))) << 5)
                           + (lane >> 4) * 8;
    v8h lo = *(const v8h*)p;
    v8h hi = *(const v8h*)(p + 16);
    v16h a = __builtin_shufflevector(lo, hi, 0,1,2,3,4,5,6,7,
                                             8,9,10,11,12,13,14,15);
    v8f z = {};
    v8f d0 = __builtin_amdgcn_wmma_f32_16x16x32_f16(false, a, false, wb[0][lane],
                                                    (short)0, z, false, false);
    v8f d1 = __builtin_amdgcn_wmma_f32_16x16x32_f16(false, a, false, wb[1][lane],
                                                    (short)0, z, false, false);
    const int n0 = lane & 15;
    #pragma unroll
    for (int i = 0; i < 8; ++i) {
        int m = 8 * (lane >> 4) + i;
        stage[wave][m][n0]      = lrelu(d0[i]);       // pw0 has NO BN scale
        stage[wave][m][n0 + 16] = lrelu(d1[i]);
    }
    __syncthreads();

    // pw1: 32 lanes = 16 pixels x 2 output channels
    int pix = lane & 15, oc = lane >> 4;
    float s = 0.0f;
    #pragma unroll
    for (int k = 0; k < 32; ++k) s += stage[wave][pix][k] * pw1[k * 2 + oc];
    disp[((size_t)(pixbase + pix)) * 2 + oc] = s;
}

// ---------------------------------------------------------------------------
// B-spline upsample of disp [4,64,64,2] to 512^2, + grid, + bilinear warp.
// Exactly mirrors the reference math (incl. u<->m pairing and eps-bilinear).
// ---------------------------------------------------------------------------
__global__ __launch_bounds__(256) void bspline_warp(
        const float* __restrict__ disp, const float* __restrict__ moving,
        float* __restrict__ out) {
    int idx = blockIdx.x * blockDim.x + threadIdx.x;  // over 4*512*512
    if (idx >= 4 * 512 * 512) return;
    int x = idx & 511, y = (idx >> 9) & 511, b = idx >> 18;

    float xx = fminf((float)x * 0.125f, 63.0f);
    float yy = fminf((float)y * 0.125f, 63.0f);
    int ii = (int)xx, jj = (int)yy;
    float u = xx * (1.0f / 64.0f);
    float v = yy * (1.0f / 64.0f);

    float u2 = u * u, u3 = u2 * u;
    float v2 = v * v, v3 = v2 * v;
    float Bu[4] = { -u3 + 3.f*u2 - 3.f*u + 1.f,  3.f*u3 - 6.f*u2 + 4.f,
                    -3.f*u3 + 3.f*u2 + 3.f*u + 1.f,  u3 };
    float Bv[4] = { -v3 + 3.f*v2 - 3.f*v + 1.f,  3.f*v3 - 6.f*v2 + 4.f,
                    -3.f*v3 + 3.f*v2 + 3.f*v + 1.f,  v3 };

    float s0 = 0.0f, s1 = 0.0f;
    #pragma unroll
    for (int m = 0; m < 4; ++m) {
        int ry = jj + m - 1;
        bool rok = (unsigned)ry < 64u;
        #pragma unroll
        for (int n = 0; n < 4; ++n) {
            int rx = ii + n - 1;
            bool ok = rok && (unsigned)rx < 64u;
            size_t gp = ((size_t)((b * 64 + (ok ? ry : 0)) * 64 + (ok ? rx : 0))) * 2;
            float g0 = ok ? disp[gp]     : 0.0f;
            float g1 = ok ? disp[gp + 1] : 0.0f;
            float w = Bu[m] * Bv[n];                  // Bu pairs with m-axis
            s0 += w * g0;  s1 += w * g1;
        }
    }
    float wx = s0 + (float)x;
    float wy = s1 + (float)y;

    // eps-bilinear, per reference
    const float eps = 1e-5f;
    float fx0 = floorf(wx), fy0 = floorf(wy);
    float x0 = fminf(fmaxf(fx0,        0.f), 511.f);
    float x1 = fminf(fmaxf(fx0 + 1.f,  0.f), 511.f);
    float y0 = fminf(fmaxf(fy0,        0.f), 511.f);
    float y1 = fminf(fmaxf(fy0 + 1.f,  0.f), 511.f);
    int x0i = (int)x0, x1i = (int)x1, y0i = (int)y0, y1i = (int)y1;
    size_t ib = (size_t)b * 512 * 512;
    float Q1 = moving[ib + (size_t)y0i * 512 + x0i];
    float Q2 = moving[ib + (size_t)y1i * 512 + x0i];
    float Q3 = moving[ib + (size_t)y0i * 512 + x1i];
    float Q4 = moving[ib + (size_t)y1i * 512 + x1i];
    float dx = x1 - x0 + eps, dy = y1 - y0 + eps;
    float R1 = (x1 - wx) / dx * Q1 + (wx - x0) / dx * Q3;
    float R2 = (x1 - wx) / dx * Q2 + (wx - x0) / dx * Q4;
    float wv = (y1 - wy) / dy * R1 + (wy - y0) / dy * R2;

    out[ib + (size_t)y * 512 + x] = wv;
    size_t gbase = (size_t)4 * 512 * 512;
    out[gbase + ((size_t)(b * 2 + 0) * 512 + y) * 512 + x] = wx;
    out[gbase + ((size_t)(b * 2 + 1) * 512 + y) * 512 + x] = wy;
}

// ---------------------------------------------------------------------------
// Workspace layout (bytes)
// ---------------------------------------------------------------------------
#define WS_X1     ((size_t)0)            // f16 [4,256,256,32]  16 MB
#define WS_X2     ((size_t)16777216)     // f16 [4,128,128,32]   4 MB
#define WS_X3     ((size_t)20971520)     // f16 [4,64,64,32]     1 MB
#define WS_F0     ((size_t)22020096)     // f16 [4,64,64,32]     1 MB
#define WS_F1     ((size_t)23068672)     // f16 [4,64,64,32]     1 MB
#define WS_DISP   ((size_t)24117248)     // f32 [4,64,64,2]    128 KB
#define WS_WP1    ((size_t)24248320)     // f16 packed w1       18 KB
#define WS_WP2    ((size_t)24266752)
#define WS_WPF0   ((size_t)24285184)
#define WS_WPF1   ((size_t)24303616)
#define WS_WPPW0  ((size_t)24322048)     //  2 KB

extern "C" void kernel_launch(void* const* d_in, const int* in_sizes, int n_in,
                              void* d_out, int out_size, void* d_ws, size_t ws_size,
                              hipStream_t stream) {
    const float* fixed  = (const float*)d_in[0];
    const float* moving = (const float*)d_in[1];
    const float* w0  = (const float*)d_in[2];
    const float* w1  = (const float*)d_in[3];
    const float* w2  = (const float*)d_in[4];
    const float* fw0 = (const float*)d_in[5];
    const float* fw1 = (const float*)d_in[6];
    const float* pw0 = (const float*)d_in[7];
    const float* pw1 = (const float*)d_in[8];
    float* out = (float*)d_out;
    char* ws = (char*)d_ws;

    _Float16* x1   = (_Float16*)(ws + WS_X1);
    _Float16* x2   = (_Float16*)(ws + WS_X2);
    _Float16* x3   = (_Float16*)(ws + WS_X3);
    _Float16* f0   = (_Float16*)(ws + WS_F0);
    _Float16* f1   = (_Float16*)(ws + WS_F1);
    float*    disp = (float*)   (ws + WS_DISP);
    _Float16* wp1  = (_Float16*)(ws + WS_WP1);
    _Float16* wp2  = (_Float16*)(ws + WS_WP2);
    _Float16* wpf0 = (_Float16*)(ws + WS_WPF0);
    _Float16* wpf1 = (_Float16*)(ws + WS_WPF1);
    _Float16* wppw = (_Float16*)(ws + WS_WPPW0);

    // 1) pack weights into WMMA B-fragment layout
    pack_weights<<<36, 256, 0, stream>>>(w1,  wp1,  9);
    pack_weights<<<36, 256, 0, stream>>>(w2,  wp2,  9);
    pack_weights<<<36, 256, 0, stream>>>(fw0, wpf0, 9);
    pack_weights<<<36, 256, 0, stream>>>(fw1, wpf1, 9);
    pack_weights<<<4,  256, 0, stream>>>(pw0, wppw, 1);

    // 2) stage 1: 2->32 conv + pool (VALU; K=18 too thin for WMMA)
    conv1_pool<<<(4 * 256 * 256) / 256, 256, 0, stream>>>(fixed, moving, w0, x1);

    // 3) WMMA conv stages (LDS-staged halo tiles)
    conv3x3_wmma<true ><<<dim3(16, 16, 4), 256, 0, stream>>>(x1, wp1,  x2, 256, 256);
    conv3x3_wmma<true ><<<dim3( 8,  8, 4), 256, 0, stream>>>(x2, wp2,  x3, 128, 128);
    conv3x3_wmma<false><<<dim3( 4,  4, 4), 256, 0, stream>>>(x3, wpf0, f0,  64,  64);
    conv3x3_wmma<false><<<dim3( 4,  4, 4), 256, 0, stream>>>(f0, wpf1, f1,  64,  64);

    // 4) pointwise head -> disp
    pointwise_wmma<<<(4 * 64 * 64) / 128, 256, 0, stream>>>(f1, wppw, pw1, disp);

    // 5) B-spline upsample + grid + bilinear warp -> outputs
    bspline_warp<<<(4 * 512 * 512) / 256, 256, 0, stream>>>(disp, moving, out);
}
